// DecoderRNN_31413390803351
// MI455X (gfx1250) — matrix operand order repaired and checked by reading, compiled, not run on hardware
//
#include <hip/hip_runtime.h>
#include <hip/hip_bf16.h>
#include <math.h>

// ---------------- problem constants ----------------
#define B_  32
#define E_  512
#define H_  512
#define N_  50
#define T_  100
#define V_  32000
#define VO  (V_ - 1)          // 31999 output logits
#define VOP 32000             // VO padded to multiple of 16 (one zero weight row)
#define G_  (4 * H_)          // 2048 LSTM gates
#define K2E (2 * E_)          // 1024 concat input

typedef __attribute__((ext_vector_type(16))) _Float16 v16h;
typedef __attribute__((ext_vector_type(8)))  _Float16 v8h;
typedef __attribute__((ext_vector_type(8)))  float    v8f;

__device__ __forceinline__ float sigmoidf_(float x) { return 1.0f / (1.0f + expf(-x)); }

// ================= WMMA fragment loaders =================
// A fragment from the LDS-staged panel (row-major [32 x K], rows relative to panel).
// Lane<16: row=lane,  K = k0+{0..15}; lane>=16: row=lane-16, K = k0+{16..31}.
__device__ __forceinline__ v16h lds_a_frag(const _Float16* sA, int K,
                                           int row, int k0, int hi) {
  const _Float16* p = sA + row * K + k0 + hi * 16;
  v8h lo = *(const v8h*)p;          // K offsets +0..7
  v8h hh = *(const v8h*)(p + 8);    // K offsets +8..15
  v16h a;
#pragma unroll
  for (int i = 0; i < 8; ++i) { a[i] = lo[i]; a[i + 8] = hh[i]; }
  return a;
}

// B: W row-major [NcolsPad x K] f16 (B = W^T, 32x16 tile of K x N). Unconditional:
// W is padded so every tile's 16 columns are readable.
// N = n0 + (lane&15); lanes 0-15 hold K k0..k0+15, lanes 16-31 hold k0+16..k0+31.
__device__ __forceinline__ v16h load_b_frag(const _Float16* __restrict__ W, int K,
                                            int col, int k0, int hi) {
  return *(const v16h*)(W + (size_t)col * K + k0 + hi * 16);
}

// ================= generic skinny WMMA GEMM =================
// Y[m, n] = act( sum_k A1[m,k] W1[n,k] + sum_k A2[m,k] W2[n,k] + bias1[n] + bias2[n] )
// One wave computes a 32x16 tile. A panels are staged block-wide into LDS.
// Requires: K1, K2 multiples of 64; W matrices have NcolsPad (mult of 16) rows.
// act: 0=none, 1=tanh, 2=relu.  Optional f16 mirror in Yh.
__global__ void wmma_gemm_kernel(const _Float16* __restrict__ A1,
                                 const _Float16* __restrict__ W1, int K1,
                                 const _Float16* __restrict__ A2,
                                 const _Float16* __restrict__ W2, int K2,
                                 const float* __restrict__ bias1,
                                 const float* __restrict__ bias2,
                                 float* __restrict__ Y,
                                 _Float16* __restrict__ Yh,
                                 int NcolsPad, int Ncols, int ldY, int ldYh, int act) {
  extern __shared__ _Float16 smem[];           // [32*K1] (+ [32*K2] if A2)
  const int m0 = blockIdx.y * 32;

  // ---- block-cooperative staging of the A panel(s) into LDS ----
  {
    const int na1 = 32 * K1;
    const _Float16* src1 = A1 + (size_t)m0 * K1;   // rows m0..m0+31 are contiguous
    for (int i = threadIdx.x * 8; i < na1; i += blockDim.x * 8)
      *(v8h*)(smem + i) = *(const v8h*)(src1 + i);
    if (A2 != nullptr) {
      const int na2 = 32 * K2;
      const _Float16* src2 = A2 + (size_t)m0 * K2;
      _Float16* d2 = smem + na1;
      for (int i = threadIdx.x * 8; i < na2; i += blockDim.x * 8)
        *(v8h*)(d2 + i) = *(const v8h*)(src2 + i);
    }
  }
  __syncthreads();

  const int wavesPerBlock = blockDim.x >> 5;
  const int waveId = blockIdx.x * wavesPerBlock + (threadIdx.x >> 5);
  const int n0 = waveId * 16;
  if (n0 >= NcolsPad) return;                // wave-uniform; EXEC all-ones at WMMA
  const int lane = threadIdx.x & 31;
  const int hi   = lane >> 4;
  const int rlo  = lane & 15;
  const int col  = n0 + rlo;                 // always loadable (padded W)

  v8f accL = {};
  v8f accH = {};

  // ---- K1 loop, 64 per iteration (4 WMMAs) ----
  for (int k0 = 0; k0 < K1; k0 += 64) {
    if (k0 + 64 < K1)                        // uniform condition -> no exec churn
      __builtin_prefetch(W1 + (size_t)col * K1 + k0 + 64, 0, 3);  // WGP-scope prefetch
    v16h b0  = load_b_frag(W1, K1, col, k0,      hi);
    v16h b1  = load_b_frag(W1, K1, col, k0 + 32, hi);
    v16h aL0 = lds_a_frag(smem, K1, rlo,      k0,      hi);
    v16h aH0 = lds_a_frag(smem, K1, rlo + 16, k0,      hi);
    v16h aL1 = lds_a_frag(smem, K1, rlo,      k0 + 32, hi);
    v16h aH1 = lds_a_frag(smem, K1, rlo + 16, k0 + 32, hi);
    accL = __builtin_amdgcn_wmma_f32_16x16x32_f16(false, aL0, false, b0, (short)0, accL, false, false);
    accH = __builtin_amdgcn_wmma_f32_16x16x32_f16(false, aH0, false, b0, (short)0, accH, false, false);
    accL = __builtin_amdgcn_wmma_f32_16x16x32_f16(false, aL1, false, b1, (short)0, accL, false, false);
    accH = __builtin_amdgcn_wmma_f32_16x16x32_f16(false, aH1, false, b1, (short)0, accH, false, false);
  }
  // ---- optional second input (fused dual GEMM, e.g. LSTM x/h) ----
  if (A2 != nullptr) {
    const _Float16* sA2 = smem + 32 * K1;
    for (int k0 = 0; k0 < K2; k0 += 64) {
      v16h b0  = load_b_frag(W2, K2, col, k0,      hi);
      v16h b1  = load_b_frag(W2, K2, col, k0 + 32, hi);
      v16h aL0 = lds_a_frag(sA2, K2, rlo,      k0,      hi);
      v16h aH0 = lds_a_frag(sA2, K2, rlo + 16, k0,      hi);
      v16h aL1 = lds_a_frag(sA2, K2, rlo,      k0 + 32, hi);
      v16h aH1 = lds_a_frag(sA2, K2, rlo + 16, k0 + 32, hi);
      accL = __builtin_amdgcn_wmma_f32_16x16x32_f16(false, aL0, false, b0, (short)0, accL, false, false);
      accH = __builtin_amdgcn_wmma_f32_16x16x32_f16(false, aH0, false, b0, (short)0, accH, false, false);
      accL = __builtin_amdgcn_wmma_f32_16x16x32_f16(false, aL1, false, b1, (short)0, accL, false, false);
      accH = __builtin_amdgcn_wmma_f32_16x16x32_f16(false, aH1, false, b1, (short)0, accH, false, false);
    }
  }

  // ---- epilogue: bias + activation + store (only real columns) ----
  if (col < Ncols) {
    float bsum = 0.0f;
    if (bias1) bsum += bias1[col];
    if (bias2) bsum += bias2[col];
#pragma unroll
    for (int r = 0; r < 8; ++r) {
      // C/D layout: VGPR r, lane<16 -> M=r, lane>=16 -> M=r+8
      int rowL = m0 + r + hi * 8;
      int rowH = rowL + 16;
      float vL = accL[r] + bsum;
      float vH = accH[r] + bsum;
      if (act == 1)      { vL = tanhf(vL);      vH = tanhf(vH); }
      else if (act == 2) { vL = fmaxf(vL, 0.f); vH = fmaxf(vH, 0.f); }
      Y[(size_t)rowL * ldY + col] = vL;
      Y[(size_t)rowH * ldY + col] = vH;
      if (Yh) {
        Yh[(size_t)rowL * ldYh + col] = (_Float16)vL;
        Yh[(size_t)rowH * ldYh + col] = (_Float16)vH;
      }
    }
  }
}

// ================= setup kernels (one-time per call) =================
__global__ void f32_to_f16_kernel(const float* __restrict__ src, _Float16* __restrict__ dst, int n) {
  int i = blockIdx.x * blockDim.x + threadIdx.x;
  if (i < n) dst[i] = (_Float16)src[i];
}

__global__ void f16_zero_kernel(_Float16* __restrict__ dst, int n) {
  int i = blockIdx.x * blockDim.x + threadIdx.x;
  if (i < n) dst[i] = (_Float16)0.f;
}

// features [B,E,N] f32 -> At [(B*N) x E] f16  (rows = (b,n) pairs)
__global__ void transpose_feats_kernel(const float* __restrict__ features, _Float16* __restrict__ At) {
  int idx = blockIdx.x * blockDim.x + threadIdx.x;   // over B*N*E
  if (idx >= B_ * N_ * E_) return;
  int bn = idx >> 9;            // /E_
  int e  = idx & (E_ - 1);
  int b  = bn / N_;
  int n  = bn - b * N_;
  At[idx] = (_Float16)features[((size_t)b * E_ + e) * N_ + n];
}

// inp0[b,e] = mean_n features[b,e,n]
__global__ void avg_feats_kernel(const float* __restrict__ features, float* __restrict__ inp) {
  int idx = blockIdx.x * blockDim.x + threadIdx.x;   // over B*E
  if (idx >= B_ * E_) return;
  const float* p = features + (size_t)idx * N_;
  float s = 0.f;
#pragma unroll 10
  for (int n = 0; n < N_; ++n) s += p[n];
  inp[idx] = s * (1.0f / N_);
}

// emb_h[b,t,e] = f16(table[captions[b,t], e])
__global__ void embed_gather_kernel(const int* __restrict__ captions,
                                    const float* __restrict__ table,
                                    _Float16* __restrict__ emb_h) {
  int idx = blockIdx.x * blockDim.x + threadIdx.x;   // over B*T*E
  if (idx >= B_ * T_ * E_) return;
  int bt = idx >> 9;
  int e  = idx & (E_ - 1);
  int tok = captions[bt];
  emb_h[idx] = (_Float16)table[(size_t)tok * E_ + e];
}

__global__ void init_state_kernel(float* __restrict__ h, float* __restrict__ c,
                                  _Float16* __restrict__ h_h) {
  int idx = blockIdx.x * blockDim.x + threadIdx.x;   // over B*H
  if (idx >= B_ * H_) return;
  h[idx] = 0.f; c[idx] = 0.f; h_h[idx] = (_Float16)0.f;
}

// ================= per-step kernels =================
// x_h[b, 0:E] = f16(inp[b]); x_h[b, E:2E] = emb_h[b, t]
__global__ void concat_x_kernel(const float* __restrict__ inp,
                                const _Float16* __restrict__ emb_h,
                                _Float16* __restrict__ x_h, int t) {
  int idx = blockIdx.x * blockDim.x + threadIdx.x;   // over B*2E
  if (idx >= B_ * K2E) return;
  int b = idx >> 10;
  int j = idx & (K2E - 1);
  x_h[idx] = (j < E_) ? (_Float16)inp[(size_t)b * E_ + j]
                      : emb_h[((size_t)b * T_ + t) * E_ + (j - E_)];
}

// gates [B, 4H] in order i,f,g,o (PyTorch) -> update c, h; mirror h to f16
__global__ void lstm_update_kernel(const float* __restrict__ gates,
                                   float* __restrict__ c, float* __restrict__ h,
                                   _Float16* __restrict__ h_h) {
  int idx = blockIdx.x * blockDim.x + threadIdx.x;   // over B*H
  if (idx >= B_ * H_) return;
  int b = idx >> 9;
  int j = idx & (H_ - 1);
  const float* g = gates + (size_t)b * G_;
  float ii = g[j], ff = g[H_ + j], gg = g[2 * H_ + j], oo = g[3 * H_ + j];
  float cs = sigmoidf_(ff) * c[idx] + sigmoidf_(ii) * tanhf(gg);
  float hs = sigmoidf_(oo) * tanhf(cs);
  c[idx] = cs; h[idx] = hs; h_h[idx] = (_Float16)hs;
}

// additive attention: scores -> softmax -> context; one block per batch element
__global__ void attention_kernel(const float* __restrict__ feat_emb_t,   // [(B*N) x H]
                                 const float* __restrict__ feats_lin_t,  // [(B*N) x H]
                                 const float* __restrict__ he,           // [B x H]
                                 const float* __restrict__ hl,           // [B x H]
                                 const float* __restrict__ Wa,           // [H]
                                 const float* __restrict__ ba,           // [1]
                                 float* __restrict__ v, _Float16* __restrict__ v_h) {
  __shared__ float red[256];
  __shared__ float sc[64];
  const int b = blockIdx.x;
  const int tid = threadIdx.x;
  const float* heb = he + (size_t)b * H_;
  const float* hlb = hl + (size_t)b * H_;

  // scores[n] = sum_h Wa[h]*tanh(feat_emb[b,h,n] + he[b,h]) + ba
  for (int n = 0; n < N_; ++n) {
    const float* fe = feat_emb_t + ((size_t)b * N_ + n) * H_;
    float p = 0.f;
    for (int hh = tid; hh < H_; hh += blockDim.x)
      p += Wa[hh] * tanhf(fe[hh] + heb[hh]);
    red[tid] = p;
    __syncthreads();
    for (int s = blockDim.x >> 1; s > 0; s >>= 1) {
      if (tid < s) red[tid] += red[tid + s];
      __syncthreads();
    }
    if (tid == 0) sc[n] = red[0] + ba[0];
    __syncthreads();
  }
  // softmax over N
  if (tid == 0) {
    float mx = sc[0];
    for (int n = 1; n < N_; ++n) mx = fmaxf(mx, sc[n]);
    float sum = 0.f;
    for (int n = 0; n < N_; ++n) { sc[n] = expf(sc[n] - mx); sum += sc[n]; }
    float inv = 1.0f / sum;
    for (int n = 0; n < N_; ++n) sc[n] *= inv;
  }
  __syncthreads();
  // v[b,h] = sum_n att[n]*feats_lin[b,h,n] + hl[b,h]
  for (int hh = tid; hh < H_; hh += blockDim.x) {
    float s = hlb[hh];
    for (int n = 0; n < N_; ++n)
      s += sc[n] * feats_lin_t[((size_t)b * N_ + n) * H_ + hh];
    v[(size_t)b * H_ + hh] = s;
    v_h[(size_t)b * H_ + hh] = (_Float16)s;
  }
}

// per-row argmax over VO logits; write float(index) as sampled id
__global__ void argmax_kernel(const float* __restrict__ logits, int ldRow,
                              float* __restrict__ ids, int t) {
  __shared__ float sv[256];
  __shared__ int   si[256];
  const int b = blockIdx.x;
  const int tid = threadIdx.x;
  const float* row = logits + (size_t)b * ldRow;
  float best = -3.4e38f; int bi = 0;
  for (int i = tid; i < VO; i += blockDim.x) {
    float x = row[i];
    if (x > best) { best = x; bi = i; }
  }
  sv[tid] = best; si[tid] = bi;
  __syncthreads();
  for (int s = blockDim.x >> 1; s > 0; s >>= 1) {
    if (tid < s) {
      if (sv[tid + s] > sv[tid] || (sv[tid + s] == sv[tid] && si[tid + s] < si[tid])) {
        sv[tid] = sv[tid + s]; si[tid] = si[tid + s];
      }
    }
    __syncthreads();
  }
  if (tid == 0) ids[(size_t)b * T_ + t] = (float)si[0];
}

// ================= host orchestration =================
extern "C" void kernel_launch(void* const* d_in, const int* in_sizes, int n_in,
                              void* d_out, int out_size, void* d_ws, size_t ws_size,
                              hipStream_t stream) {
  const float* features = (const float*)d_in[0];
  const int*   captions = (const int*)  d_in[2];
  const float* table    = (const float*)d_in[3];
  const float* W_ih = (const float*)d_in[4];  const float* b_ih = (const float*)d_in[5];
  const float* W_hh = (const float*)d_in[6];  const float* b_hh = (const float*)d_in[7];
  const float* Wf   = (const float*)d_in[8];  const float* bf   = (const float*)d_in[9];
  const float* We   = (const float*)d_in[10]; const float* be   = (const float*)d_in[11];
  const float* Wlh  = (const float*)d_in[12]; const float* blh  = (const float*)d_in[13];
  const float* Weh  = (const float*)d_in[14]; const float* beh  = (const float*)d_in[15];
  const float* Wa   = (const float*)d_in[16]; const float* ba   = (const float*)d_in[17];
  const float* Wlo  = (const float*)d_in[18]; const float* blo  = (const float*)d_in[19];
  const float* Wout = (const float*)d_in[20]; const float* bout = (const float*)d_in[21];
  float* out = (float*)d_out;                       // [B,T,VO] logits then [B,T] ids
  float* ids = out + (size_t)B_ * T_ * VO;

  // ---- bump allocator over d_ws ----
  char* ws = (char*)d_ws;
  size_t off = 0;
  auto alloc = [&](size_t bytes) -> void* {
    off = (off + 255) & ~(size_t)255;
    void* p = ws + off;
    off += bytes;
    return p;
  };
  // f16 weight copies (all L2-resident: Wout_h 32.8 MB of 192 MB)
  _Float16* W_ih_h = (_Float16*)alloc((size_t)G_ * K2E * 2);
  _Float16* W_hh_h = (_Float16*)alloc((size_t)G_ * H_ * 2);
  _Float16* Wf_h   = (_Float16*)alloc((size_t)H_ * E_ * 2);
  _Float16* We_h   = (_Float16*)alloc((size_t)H_ * H_ * 2);
  _Float16* Wlh_h  = (_Float16*)alloc((size_t)H_ * H_ * 2);
  _Float16* Weh_h  = (_Float16*)alloc((size_t)H_ * H_ * 2);
  _Float16* Wlo_h  = (_Float16*)alloc((size_t)H_ * H_ * 2);
  _Float16* Wout_h = (_Float16*)alloc((size_t)VOP * H_ * 2);   // padded to 32000 rows
  // activations / state
  _Float16* At           = (_Float16*)alloc((size_t)B_ * N_ * E_ * 2);  // features^T
  float*    feats_lin_t  = (float*)   alloc((size_t)B_ * N_ * H_ * 4);
  _Float16* feats_lin_th = (_Float16*)alloc((size_t)B_ * N_ * H_ * 2);
  float*    feat_emb_t   = (float*)   alloc((size_t)B_ * N_ * H_ * 4);
  _Float16* emb_h        = (_Float16*)alloc((size_t)B_ * T_ * E_ * 2);
  float* h    = (float*)alloc((size_t)B_ * H_ * 4);
  float* c    = (float*)alloc((size_t)B_ * H_ * 4);
  float* inp  = (float*)alloc((size_t)B_ * E_ * 4);   // input_feat (vout feeds back here)
  float* hl   = (float*)alloc((size_t)B_ * H_ * 4);
  float* he   = (float*)alloc((size_t)B_ * H_ * 4);
  float* v    = (float*)alloc((size_t)B_ * H_ * 4);
  float* gates= (float*)alloc((size_t)B_ * G_ * 4);
  _Float16* h_h  = (_Float16*)alloc((size_t)B_ * H_ * 2);
  _Float16* x_h  = (_Float16*)alloc((size_t)B_ * K2E * 2);
  _Float16* hl_h = (_Float16*)alloc((size_t)B_ * H_ * 2);
  _Float16* v_h  = (_Float16*)alloc((size_t)B_ * H_ * 2);
  _Float16* vout_h = (_Float16*)alloc((size_t)B_ * H_ * 2);

  const int TB = 256;
  auto cdiv = [](int a, int b) { return (a + b - 1) / b; };

  // ---- one-time: f16 weight conversion (+ zero pad row of Wout) ----
  f32_to_f16_kernel<<<cdiv(G_ * K2E, TB), TB, 0, stream>>>(W_ih, W_ih_h, G_ * K2E);
  f32_to_f16_kernel<<<cdiv(G_ * H_,  TB), TB, 0, stream>>>(W_hh, W_hh_h, G_ * H_);
  f32_to_f16_kernel<<<cdiv(H_ * E_,  TB), TB, 0, stream>>>(Wf,   Wf_h,   H_ * E_);
  f32_to_f16_kernel<<<cdiv(H_ * H_,  TB), TB, 0, stream>>>(We,   We_h,   H_ * H_);
  f32_to_f16_kernel<<<cdiv(H_ * H_,  TB), TB, 0, stream>>>(Wlh,  Wlh_h,  H_ * H_);
  f32_to_f16_kernel<<<cdiv(H_ * H_,  TB), TB, 0, stream>>>(Weh,  Weh_h,  H_ * H_);
  f32_to_f16_kernel<<<cdiv(H_ * H_,  TB), TB, 0, stream>>>(Wlo,  Wlo_h,  H_ * H_);
  f32_to_f16_kernel<<<cdiv(VO * H_,  TB), TB, 0, stream>>>(Wout, Wout_h, VO * H_);
  f16_zero_kernel<<<cdiv((VOP - VO) * H_, TB), TB, 0, stream>>>(
      Wout_h + (size_t)VO * H_, (VOP - VO) * H_);

  // ---- one-time: hoisted attention projections ----
  transpose_feats_kernel<<<cdiv(B_ * N_ * E_, TB), TB, 0, stream>>>(features, At);
  embed_gather_kernel<<<cdiv(B_ * T_ * E_, TB), TB, 0, stream>>>(captions, table, emb_h);
  avg_feats_kernel<<<cdiv(B_ * E_, TB), TB, 0, stream>>>(features, inp);
  init_state_kernel<<<cdiv(B_ * H_, TB), TB, 0, stream>>>(h, c, h_h);

  const size_t smemH  = (size_t)32 * H_ * 2;            // 32 KB  (single A, K=512)
  const size_t smemLS = (size_t)32 * (K2E + H_) * 2;    // 96 KB  (dual A, LSTM gates)

  // feats_lin_t[(b,n), h] = relu(At @ Wf^T + bf) ; 1600 rows -> grid.y = 50
  wmma_gemm_kernel<<<dim3(H_ / 16 / 8, (B_ * N_) / 32), TB, (size_t)32 * E_ * 2, stream>>>(
      At, Wf_h, E_, nullptr, nullptr, 0, bf, nullptr,
      feats_lin_t, feats_lin_th, H_, H_, H_, H_, /*relu*/2);
  // feat_emb_t = feats_lin @ We^T + be
  wmma_gemm_kernel<<<dim3(H_ / 16 / 8, (B_ * N_) / 32), TB, smemH, stream>>>(
      feats_lin_th, We_h, H_, nullptr, nullptr, 0, be, nullptr,
      feat_emb_t, nullptr, H_, H_, H_, 0, /*none*/0);

  // ---- sequential decode: T steps ----
  const int ldOut = T_ * VO;   // row stride of a batch element inside out[B,T,VO]
  for (int t = 0; t < T_; ++t) {
    concat_x_kernel<<<cdiv(B_ * K2E, TB), TB, 0, stream>>>(inp, emb_h, x_h, t);

    // gates = x @ W_ih^T + h @ W_hh^T + b_ih + b_hh     [32 x 2048]
    wmma_gemm_kernel<<<dim3(G_ / 16 / 8, 1), TB, smemLS, stream>>>(
        x_h, W_ih_h, K2E, h_h, W_hh_h, H_, b_ih, b_hh,
        gates, nullptr, G_, G_, G_, 0, 0);

    lstm_update_kernel<<<cdiv(B_ * H_, TB), TB, 0, stream>>>(gates, c, h, h_h);

    // hl = tanh(h @ Wlh^T + blh)
    wmma_gemm_kernel<<<dim3(H_ / 16 / 8, 1), TB, smemH, stream>>>(
        h_h, Wlh_h, H_, nullptr, nullptr, 0, blh, nullptr,
        hl, hl_h, H_, H_, H_, H_, /*tanh*/1);
    // he = hl @ Weh^T + beh
    wmma_gemm_kernel<<<dim3(H_ / 16 / 8, 1), TB, smemH, stream>>>(
        hl_h, Weh_h, H_, nullptr, nullptr, 0, beh, nullptr,
        he, nullptr, H_, H_, H_, 0, 0);

    attention_kernel<<<B_, TB, 0, stream>>>(feat_emb_t, feats_lin_t, he, hl,
                                            Wa, ba, v, v_h);

    // vout = tanh(v @ Wlo^T + blo)   -> becomes next input_feat (written into inp)
    wmma_gemm_kernel<<<dim3(H_ / 16 / 8, 1), TB, smemH, stream>>>(
        v_h, Wlo_h, H_, nullptr, nullptr, 0, blo, nullptr,
        inp, vout_h, H_, H_, H_, H_, /*tanh*/1);

    // logits[b, t, :] = vout @ Wout^T + bout   [32 x 31999]; Wout_h padded to 32000,
    // streamed from L2-resident f16 weights
    wmma_gemm_kernel<<<dim3(VOP / 16 / 8, 1), TB, smemH, stream>>>(
        vout_h, Wout_h, H_, nullptr, nullptr, 0, bout, nullptr,
        out + (size_t)t * VO, nullptr, VOP, VO, ldOut, 0, 0);

    argmax_kernel<<<B_, TB, 0, stream>>>(out + (size_t)t * VO, ldOut, ids, t);
  }
}
